// MultiheadAttention_18665927868973
// MI455X (gfx1250) — matrix operand order repaired and checked by reading, not compiled, not run on hardware
//
#include <hip/hip_runtime.h>

// ---------------- constants ----------------
#define S_LEN   1024
#define BATCH   16
#define NHEADS  8
#define HDIM    32
#define QDIM    128
#define FFDIM   512
#define NTOK    (S_LEN * BATCH)   // 16384
#define LOG2E   1.4426950408889634f

typedef __attribute__((ext_vector_type(16))) _Float16 v16h;
typedef __attribute__((ext_vector_type(8)))  float    v8f;

union FragU { float4 f4[2]; v16h h; };

// A-fragment (16x32 f16): lane L holds row M=L&15,
// K = k0 + hi*16 + 0..15 contiguous. Caller passes p = rowbase + k0 + hi*16.
__device__ __forceinline__ v16h ldA16(const _Float16* p) {
    FragU u;
    u.f4[0] = *(const float4*)(p);
    u.f4[1] = *(const float4*)(p + 8);
    return u.h;
}

// B-fragment (32x16 f16, stored as [N][K]): lane L holds col N=L&15,
// K = k0 + hi*16 + 0..15 contiguous. Caller passes p = colbase + k0 + hi*16.
__device__ __forceinline__ v16h ldB16(const _Float16* p) {
    FragU u;
    u.f4[0] = *(const float4*)(p);
    u.f4[1] = *(const float4*)(p + 8);
    return u.h;
}

__device__ __forceinline__ v8f wmma32(v16h a, v16h b, v8f c) {
    return __builtin_amdgcn_wmma_f32_16x16x32_f16(false, a, false, b,
                                                  (short)0, c, false, false);
}

__device__ __forceinline__ void wait_ds0() {
#if __has_builtin(__builtin_amdgcn_s_wait_dscnt)
    __builtin_amdgcn_s_wait_dscnt(0);
#else
    asm volatile("s_wait_dscnt 0" ::: "memory");
#endif
}

// ---------------- prep kernels ----------------

// Xh[b][s][q] = (f16) sequence[s][b][q]
__global__ __launch_bounds__(256) void cast_x_kernel(const float* __restrict__ seq,
                                                     _Float16* __restrict__ Xh) {
    int i = blockIdx.x * 256 + threadIdx.x;          // [0, NTOK*QDIM)
    int q = i & (QDIM - 1);
    int rest = i >> 7;                                // b*S + s
    int s = rest & (S_LEN - 1);
    int b = rest >> 10;
    Xh[i] = (_Float16)seq[((size_t)s * BATCH + b) * QDIM + q];
}

// dst[batch][n][k] = (f16) src[batch][k][n]   (weight transpose to [N][K] f16)
__global__ __launch_bounds__(256) void transpose_w_kernel(const float* __restrict__ src,
                                                          _Float16* __restrict__ dst,
                                                          int K, int N) {
    int i = blockIdx.x * 256 + threadIdx.x;
    if (i >= K * N) return;
    int k = i % K, n = i / K;
    size_t base = (size_t)blockIdx.y * K * N;
    dst[base + i] = (_Float16)src[base + (size_t)k * N + n];
}

// ---------------- QKV projection ----------------
// grid: (S/128, B*H).  Wave w handles 16 rows. Out: Qh/Kh [bh][s][32], Vt [bh][32][s]
__global__ __launch_bounds__(256) void qkv_kernel(
    const _Float16* __restrict__ Xh,
    const _Float16* __restrict__ WqT, const _Float16* __restrict__ WkT,
    const _Float16* __restrict__ WvT,
    const float* __restrict__ bq, const float* __restrict__ bk,
    const float* __restrict__ bv,
    _Float16* __restrict__ Qh, _Float16* __restrict__ Kh,
    _Float16* __restrict__ Vt) {
    const int lane = threadIdx.x & 31, wave = threadIdx.x >> 5;
    const int m = lane & 15, hi = lane >> 4;
    const int bh = blockIdx.y, b = bh >> 3, h = bh & 7;
    const int s0 = blockIdx.x * 128 + wave * 16;

    const _Float16* Arow = Xh + ((size_t)(b * S_LEN + s0 + m)) * QDIM + hi * 16;
    const _Float16* wq = WqT + (size_t)h * HDIM * QDIM;
    const _Float16* wk = WkT + (size_t)h * HDIM * QDIM;
    const _Float16* wv = WvT + (size_t)h * HDIM * QDIM;

    v8f cq0{}, cq1{}, ck0{}, ck1{}, cv0{}, cv1{};
#pragma unroll
    for (int k = 0; k < QDIM; k += 32) {
        v16h a = ldA16(Arow + k);
        cq0 = wmma32(a, ldB16(wq + (size_t)m * QDIM + k + hi * 16), cq0);
        cq1 = wmma32(a, ldB16(wq + (size_t)(16 + m) * QDIM + k + hi * 16), cq1);
        ck0 = wmma32(a, ldB16(wk + (size_t)m * QDIM + k + hi * 16), ck0);
        ck1 = wmma32(a, ldB16(wk + (size_t)(16 + m) * QDIM + k + hi * 16), ck1);
        cv0 = wmma32(a, ldB16(wv + (size_t)m * QDIM + k + hi * 16), cv0);
        cv1 = wmma32(a, ldB16(wv + (size_t)(16 + m) * QDIM + k + hi * 16), cv1);
    }
    const float bq0 = bq[h * HDIM + m], bq1 = bq[h * HDIM + 16 + m];
    const float bk0 = bk[h * HDIM + m], bk1 = bk[h * HDIM + 16 + m];
    const float bv0 = bv[h * HDIM + m], bv1 = bv[h * HDIM + 16 + m];
#pragma unroll
    for (int r = 0; r < 8; r++) {
        const int row = s0 + r + 8 * hi;
        size_t qb = ((size_t)bh * S_LEN + row) * HDIM;
        Qh[qb + m]      = (_Float16)(cq0[r] + bq0);
        Qh[qb + 16 + m] = (_Float16)(cq1[r] + bq1);
        Kh[qb + m]      = (_Float16)(ck0[r] + bk0);
        Kh[qb + 16 + m] = (_Float16)(ck1[r] + bk1);
        Vt[((size_t)bh * HDIM + m) * S_LEN + row]      = (_Float16)(cv0[r] + bv0);
        Vt[((size_t)bh * HDIM + 16 + m) * S_LEN + row] = (_Float16)(cv1[r] + bv1);
    }
}

// ---------------- flash attention ----------------
// grid: (S/128, B*H). Wave = 16 Q rows; online softmax over 32-key tiles.
__global__ __launch_bounds__(256) void attn_kernel(
    const _Float16* __restrict__ Qh, const _Float16* __restrict__ Kh,
    const _Float16* __restrict__ Vt, _Float16* __restrict__ Oh) {
    __shared__ _Float16 Pt[8][16 * 48];   // per-wave 16x32 P tile, 48-half row stride

    const int lane = threadIdx.x & 31, wave = threadIdx.x >> 5;
    const int m = lane & 15, hi = lane >> 4;
    const int bh = blockIdx.y, b = bh >> 3, h = bh & 7;
    const int s0 = blockIdx.x * 128 + wave * 16;

    v16h qa = ldA16(Qh + ((size_t)bh * S_LEN + s0 + m) * HDIM + hi * 16);
    const float sc = 0.17677669529663687f * LOG2E;   // 1/sqrt(32) folded with log2(e)

    v8f o0{}, o1{};
    float mrow[8], lrow[8];
#pragma unroll
    for (int r = 0; r < 8; r++) { mrow[r] = -1e30f; lrow[r] = 0.0f; }

    _Float16* prow = &Pt[wave][0];
    const _Float16* kbase = Kh + (size_t)bh * S_LEN * HDIM;
    const _Float16* vbase = Vt + (size_t)bh * HDIM * S_LEN;

    for (int t = 0; t < S_LEN; t += 32) {
        v16h kb0 = ldB16(kbase + (size_t)(t + m) * HDIM + hi * 16);
        v16h kb1 = ldB16(kbase + (size_t)(t + 16 + m) * HDIM + hi * 16);
        v8f z0{}, z1{};
        z0 = wmma32(qa, kb0, z0);
        z1 = wmma32(qa, kb1, z1);

        float p0[8], p1[8];
#pragma unroll
        for (int r = 0; r < 8; r++) {
            float x0 = z0[r] * sc, x1 = z1[r] * sc;
            float tm = fmaxf(x0, x1);
            tm = fmaxf(tm, __shfl_xor(tm, 1, 16));
            tm = fmaxf(tm, __shfl_xor(tm, 2, 16));
            tm = fmaxf(tm, __shfl_xor(tm, 4, 16));
            tm = fmaxf(tm, __shfl_xor(tm, 8, 16));
            float mn = fmaxf(mrow[r], tm);
            float al = exp2f(mrow[r] - mn);
            p0[r] = exp2f(x0 - mn);
            p1[r] = exp2f(x1 - mn);
            lrow[r] = lrow[r] * al + p0[r] + p1[r];
            mrow[r] = mn;
            o0[r] *= al;
            o1[r] *= al;
        }
        // relayout P (C-frag) -> A-frag via LDS
#pragma unroll
        for (int r = 0; r < 8; r++) {
            int row = r + 8 * hi;
            prow[row * 48 + m]      = (_Float16)p0[r];
            prow[row * 48 + 16 + m] = (_Float16)p1[r];
        }
        wait_ds0();
        v16h pa  = ldA16(prow + m * 48 + hi * 16);
        v16h vb0 = ldB16(vbase + (size_t)m * S_LEN + t + hi * 16);
        v16h vb1 = ldB16(vbase + (size_t)(16 + m) * S_LEN + t + hi * 16);
        o0 = wmma32(pa, vb0, o0);
        o1 = wmma32(pa, vb1, o1);
    }
#pragma unroll
    for (int r = 0; r < 8; r++) {
        float l = lrow[r];
        l += __shfl_xor(l, 1, 16);
        l += __shfl_xor(l, 2, 16);
        l += __shfl_xor(l, 4, 16);
        l += __shfl_xor(l, 8, 16);
        float inv = 1.0f / l;
        const int row = s0 + r + 8 * hi;
        size_t ob = ((size_t)b * S_LEN + row) * (NHEADS * HDIM) + h * HDIM;
        Oh[ob + m]      = (_Float16)(o0[r] * inv);
        Oh[ob + 16 + m] = (_Float16)(o1[r] * inv);
    }
}

// ---------------- Wo projection + residual + layernorm ----------------
// grid: NTOK/128 blocks. Rows indexed as b*S + s.
__global__ __launch_bounds__(256) void proj_ln_kernel(
    const _Float16* __restrict__ Oh, const _Float16* __restrict__ WoT,
    const float* __restrict__ bo, const float* __restrict__ seq,
    const float* __restrict__ gamma, const float* __restrict__ beta,
    float* __restrict__ attf, _Float16* __restrict__ atth) {
    const int lane = threadIdx.x & 31, wave = threadIdx.x >> 5;
    const int m = lane & 15, hi = lane >> 4;
    const int row0 = blockIdx.x * 128 + wave * 16;
    const int KD = NHEADS * HDIM;  // 256

    const _Float16* Arow = Oh + (size_t)(row0 + m) * KD + hi * 16;
    v8f acc[8] = {};
#pragma unroll
    for (int k = 0; k < KD; k += 32) {
        v16h a = ldA16(Arow + k);
#pragma unroll
        for (int nt = 0; nt < 8; nt++)
            acc[nt] = wmma32(a, ldB16(WoT + (size_t)(nt * 16 + m) * KD + k + hi * 16), acc[nt]);
    }
    float bov[8], g[8], bt[8];
#pragma unroll
    for (int nt = 0; nt < 8; nt++) {
        int n = nt * 16 + m;
        bov[nt] = bo[n]; g[nt] = gamma[n]; bt[nt] = beta[n];
    }
#pragma unroll
    for (int r = 0; r < 8; r++) {
        const int row = row0 + r + 8 * hi;
        const int s = row & (S_LEN - 1), bb = row >> 10;
        const float* xr = seq + ((size_t)s * BATCH + bb) * QDIM;
        float v[8]; float sum = 0.0f;
#pragma unroll
        for (int nt = 0; nt < 8; nt++) {
            v[nt] = acc[nt][r] + bov[nt] + xr[nt * 16 + m];
            sum += v[nt];
        }
        sum += __shfl_xor(sum, 1, 16);
        sum += __shfl_xor(sum, 2, 16);
        sum += __shfl_xor(sum, 4, 16);
        sum += __shfl_xor(sum, 8, 16);
        float mu = sum * (1.0f / QDIM);
        float var = 0.0f;
#pragma unroll
        for (int nt = 0; nt < 8; nt++) { float d = v[nt] - mu; var += d * d; }
        var += __shfl_xor(var, 1, 16);
        var += __shfl_xor(var, 2, 16);
        var += __shfl_xor(var, 4, 16);
        var += __shfl_xor(var, 8, 16);
        float rs = rsqrtf(var * (1.0f / QDIM) + 1e-5f);
        float* orow = attf + (size_t)row * QDIM;
        _Float16* hrow = atth + (size_t)row * QDIM;
#pragma unroll
        for (int nt = 0; nt < 8; nt++) {
            float y = (v[nt] - mu) * rs * g[nt] + bt[nt];
            orow[nt * 16 + m] = y;
            hrow[nt * 16 + m] = (_Float16)y;
        }
    }
}

// ---------------- generic GEMM + bias + ReLU (f16 out) ----------------
// grid: (NTOK/128, N/128). C[M,N] = relu(A[M,K] @ WT[N,K]^T + bias)
__global__ __launch_bounds__(256) void gemm_relu_kernel(
    const _Float16* __restrict__ A, const _Float16* __restrict__ WT,
    const float* __restrict__ bias, _Float16* __restrict__ Out,
    int K, int N) {
    const int lane = threadIdx.x & 31, wave = threadIdx.x >> 5;
    const int m = lane & 15, hi = lane >> 4;
    const int row0 = blockIdx.x * 128 + wave * 16;
    const int n0 = blockIdx.y * 128;

    const _Float16* Arow = A + (size_t)(row0 + m) * K + hi * 16;
    v8f acc[8] = {};
    for (int k = 0; k < K; k += 32) {
        v16h a = ldA16(Arow + k);
#pragma unroll
        for (int nt = 0; nt < 8; nt++)
            acc[nt] = wmma32(a, ldB16(WT + (size_t)(n0 + nt * 16 + m) * K + k + hi * 16), acc[nt]);
    }
    float bv[8];
#pragma unroll
    for (int nt = 0; nt < 8; nt++) bv[nt] = bias[n0 + nt * 16 + m];
#pragma unroll
    for (int r = 0; r < 8; r++) {
        const int row = row0 + r + 8 * hi;
        _Float16* orow = Out + (size_t)row * N + n0;
#pragma unroll
        for (int nt = 0; nt < 8; nt++)
            orow[nt * 16 + m] = (_Float16)fmaxf(acc[nt][r] + bv[nt], 0.0f);
    }
}

// ---------------- FFN3 + residual + layernorm -> final output ----------------
// grid: NTOK/128. out[s][b][q] (f32), rows indexed b*S+s.
__global__ __launch_bounds__(256) void ffn3_ln_kernel(
    const _Float16* __restrict__ H2, const _Float16* __restrict__ W3T,
    const float* __restrict__ b3, const float* __restrict__ attf,
    const float* __restrict__ gamma, const float* __restrict__ beta,
    float* __restrict__ out) {
    const int lane = threadIdx.x & 31, wave = threadIdx.x >> 5;
    const int m = lane & 15, hi = lane >> 4;
    const int row0 = blockIdx.x * 128 + wave * 16;
    const int K = FFDIM;

    const _Float16* Arow = H2 + (size_t)(row0 + m) * K + hi * 16;
    v8f acc[8] = {};
    for (int k = 0; k < K; k += 32) {
        v16h a = ldA16(Arow + k);
#pragma unroll
        for (int nt = 0; nt < 8; nt++)
            acc[nt] = wmma32(a, ldB16(W3T + (size_t)(nt * 16 + m) * K + k + hi * 16), acc[nt]);
    }
    float bv[8], g[8], bt[8];
#pragma unroll
    for (int nt = 0; nt < 8; nt++) {
        int n = nt * 16 + m;
        bv[nt] = b3[n]; g[nt] = gamma[n]; bt[nt] = beta[n];
    }
#pragma unroll
    for (int r = 0; r < 8; r++) {
        const int row = row0 + r + 8 * hi;
        const int s = row & (S_LEN - 1), bb = row >> 10;
        const float* ar = attf + (size_t)row * QDIM;
        float v[8]; float sum = 0.0f;
#pragma unroll
        for (int nt = 0; nt < 8; nt++) {
            v[nt] = acc[nt][r] + bv[nt] + ar[nt * 16 + m];
            sum += v[nt];
        }
        sum += __shfl_xor(sum, 1, 16);
        sum += __shfl_xor(sum, 2, 16);
        sum += __shfl_xor(sum, 4, 16);
        sum += __shfl_xor(sum, 8, 16);
        float mu = sum * (1.0f / QDIM);
        float var = 0.0f;
#pragma unroll
        for (int nt = 0; nt < 8; nt++) { float d = v[nt] - mu; var += d * d; }
        var += __shfl_xor(var, 1, 16);
        var += __shfl_xor(var, 2, 16);
        var += __shfl_xor(var, 4, 16);
        var += __shfl_xor(var, 8, 16);
        float rs = rsqrtf(var * (1.0f / QDIM) + 1e-5f);
        float* orow = out + ((size_t)s * BATCH + bb) * QDIM;
#pragma unroll
        for (int nt = 0; nt < 8; nt++)
            orow[nt * 16 + m] = (v[nt] - mu) * rs * g[nt] + bt[nt];
    }
}

// ---------------- host launch ----------------
extern "C" void kernel_launch(void* const* d_in, const int* in_sizes, int n_in,
                              void* d_out, int out_size, void* d_ws, size_t ws_size,
                              hipStream_t stream) {
    const float* seq    = (const float*)d_in[0];
    const float* Wq     = (const float*)d_in[1];
    const float* bq     = (const float*)d_in[2];
    const float* Wk     = (const float*)d_in[3];
    const float* bk     = (const float*)d_in[4];
    const float* Wv     = (const float*)d_in[5];
    const float* bv     = (const float*)d_in[6];
    const float* Wo     = (const float*)d_in[7];
    const float* bo     = (const float*)d_in[8];
    const float* gamma1 = (const float*)d_in[9];
    const float* beta1  = (const float*)d_in[10];
    const float* W1     = (const float*)d_in[11];
    const float* b1     = (const float*)d_in[12];
    const float* W2     = (const float*)d_in[13];
    const float* b2     = (const float*)d_in[14];
    const float* W3     = (const float*)d_in[15];
    const float* b3     = (const float*)d_in[16];
    float* out = (float*)d_out;

    char* ws = (char*)d_ws;
    size_t off = 0;
    auto take = [&](size_t bytes) -> char* {
        char* p = ws + off;
        off += (bytes + 255) & ~(size_t)255;
        return p;
    };
    _Float16* Xh   = (_Float16*)take((size_t)NTOK * QDIM * 2);            // [b][s][128]
    _Float16* Qh   = (_Float16*)take((size_t)BATCH * NHEADS * S_LEN * HDIM * 2);
    _Float16* Kh   = (_Float16*)take((size_t)BATCH * NHEADS * S_LEN * HDIM * 2);
    _Float16* Vt   = (_Float16*)take((size_t)BATCH * NHEADS * HDIM * S_LEN * 2);
    _Float16* Oh   = (_Float16*)take((size_t)NTOK * NHEADS * HDIM * 2);   // [b][s][256]
    float*    attf = (float*)take((size_t)NTOK * QDIM * 4);
    _Float16* atth = (_Float16*)take((size_t)NTOK * QDIM * 2);
    _Float16* h1h  = (_Float16*)take((size_t)NTOK * FFDIM * 2);
    _Float16* h2h  = (_Float16*)take((size_t)NTOK * FFDIM * 2);
    _Float16* WqT  = (_Float16*)take((size_t)NHEADS * HDIM * QDIM * 2);
    _Float16* WkT  = (_Float16*)take((size_t)NHEADS * HDIM * QDIM * 2);
    _Float16* WvT  = (_Float16*)take((size_t)NHEADS * HDIM * QDIM * 2);
    _Float16* WoT  = (_Float16*)take((size_t)QDIM * NHEADS * HDIM * 2);
    _Float16* W1T  = (_Float16*)take((size_t)FFDIM * QDIM * 2);
    _Float16* W2T  = (_Float16*)take((size_t)FFDIM * FFDIM * 2);
    _Float16* W3T  = (_Float16*)take((size_t)QDIM * FFDIM * 2);

    // prep: cast + transpose everything to f16 [N][K]
    cast_x_kernel<<<NTOK * QDIM / 256, 256, 0, stream>>>(seq, Xh);
    transpose_w_kernel<<<dim3((QDIM * HDIM + 255) / 256, NHEADS), 256, 0, stream>>>(Wq, WqT, QDIM, HDIM);
    transpose_w_kernel<<<dim3((QDIM * HDIM + 255) / 256, NHEADS), 256, 0, stream>>>(Wk, WkT, QDIM, HDIM);
    transpose_w_kernel<<<dim3((QDIM * HDIM + 255) / 256, NHEADS), 256, 0, stream>>>(Wv, WvT, QDIM, HDIM);
    transpose_w_kernel<<<dim3((NHEADS * HDIM * QDIM + 255) / 256, 1), 256, 0, stream>>>(Wo, WoT, NHEADS * HDIM, QDIM);
    transpose_w_kernel<<<dim3((QDIM * FFDIM + 255) / 256, 1), 256, 0, stream>>>(W1, W1T, QDIM, FFDIM);
    transpose_w_kernel<<<dim3((FFDIM * FFDIM + 255) / 256, 1), 256, 0, stream>>>(W2, W2T, FFDIM, FFDIM);
    transpose_w_kernel<<<dim3((FFDIM * QDIM + 255) / 256, 1), 256, 0, stream>>>(W3, W3T, FFDIM, QDIM);

    // QKV projections
    qkv_kernel<<<dim3(S_LEN / 128, BATCH * NHEADS), 256, 0, stream>>>(
        Xh, WqT, WkT, WvT, bq, bk, bv, Qh, Kh, Vt);

    // flash attention
    attn_kernel<<<dim3(S_LEN / 128, BATCH * NHEADS), 256, 0, stream>>>(Qh, Kh, Vt, Oh);

    // Wo + residual + LN
    proj_ln_kernel<<<NTOK / 128, 256, 0, stream>>>(Oh, WoT, bo, seq, gamma1, beta1, attf, atth);

    // FFN
    gemm_relu_kernel<<<dim3(NTOK / 128, FFDIM / 128), 256, 0, stream>>>(atth, W1T, b1, h1h, QDIM, FFDIM);
    gemm_relu_kernel<<<dim3(NTOK / 128, FFDIM / 128), 256, 0, stream>>>(h1h, W2T, b2, h2h, FFDIM, FFDIM);
    ffn3_ln_kernel<<<NTOK / 128, 256, 0, stream>>>(h2h, W3T, b3, attf, gamma1, beta1, out);
}
